// ENNWithAttention_47502338294475
// MI455X (gfx1250) — matrix operand order repaired and checked by reading, not compiled, not run on hardware
//
#include <hip/hip_runtime.h>

// ---------------- CDNA5 WMMA types ----------------
typedef __bf16 v16bf __attribute__((ext_vector_type(16)));
typedef __bf16 v8bf  __attribute__((ext_vector_type(8)));
typedef float  v8f   __attribute__((ext_vector_type(8)));
typedef float  vf4   __attribute__((ext_vector_type(4)));

// Problem constants (from reference)
#define BB   64
#define NN   512
#define SS   256
#define HH   64
#define CC_  64

__device__ __forceinline__ void put4(v16bf& d, int o, vf4 v) {
  d[o]     = (__bf16)v[0];
  d[o + 1] = (__bf16)v[1];
  d[o + 2] = (__bf16)v[2];
  d[o + 3] = (__bf16)v[3];
}

// =====================================================================
// bf16-WMMA GEMM, exact-fit shapes (M%16==0, Ncol%32==0, K%32==0).
//   C[M,Ncol] = act(A[M,K] @ B + bias)
// TBT=1: B given transposed as BT[Ncol,K] (contiguous-K vector loads)
// TBT=0: B given normal    as B [K,Ncol] (unconditional scalar gathers)
// One wave computes a 16x32 output slab (two 16x16 tiles share A-frag).
// Optional batching via grid.y + strides.
// =====================================================================
template <int TBT>
__global__ __launch_bounds__(256) void gemm_wmma(
    const float* __restrict__ A, int lda,
    const float* __restrict__ Bm, int ldb,
    const float* __restrict__ bias,
    float* __restrict__ Cm, int ldc,
    int M, int Ncol, int K,
    long long sA, long long sB, long long sC, int act)
{
  int wave = threadIdx.x >> 5;
  int lane = threadIdx.x & 31;
  int tilesN2 = Ncol >> 5;
  int tilesM  = M >> 4;
  int tile = blockIdx.x * 8 + wave;
  if (tile >= tilesM * tilesN2) return;
  int tm = tile / tilesN2, tn = tile % tilesN2;

  const float* Ab = A  + sA * blockIdx.y;
  const float* Bb = Bm + sB * blockIdx.y;
  float*       Cb = Cm + sC * blockIdx.y;

  int half = lane >> 4, lr = lane & 15;
  int c0 = tn * 32 + lr;
  int c1 = c0 + 16;
  const float* arowp = Ab + (long long)(tm * 16 + lr) * lda + half * 16;
  const float* bp0   = Bb + (long long)c0 * ldb + half * 16;  // TBT=1 path
  const float* bp1   = Bb + (long long)c1 * ldb + half * 16;

  v8f acc0 = {}, acc1 = {};
  for (int kk = 0; kk < K; kk += 32) {
    // speculative prefetch of next K-chunk (dropped if OOB)
    __builtin_prefetch(arowp + kk + 32, 0, 1);

    v16bf af, bf0, bf1;
    // ---- A fragment: 16 contiguous-in-K floats per lane (4x b128) ----
    put4(af, 0,  *(const vf4*)(arowp + kk));
    put4(af, 4,  *(const vf4*)(arowp + kk + 4));
    put4(af, 8,  *(const vf4*)(arowp + kk + 8));
    put4(af, 12, *(const vf4*)(arowp + kk + 12));

    if (TBT) {
      __builtin_prefetch(bp0 + kk + 32, 0, 1);
      __builtin_prefetch(bp1 + kk + 32, 0, 1);
      put4(bf0, 0,  *(const vf4*)(bp0 + kk));
      put4(bf0, 4,  *(const vf4*)(bp0 + kk + 4));
      put4(bf0, 8,  *(const vf4*)(bp0 + kk + 8));
      put4(bf0, 12, *(const vf4*)(bp0 + kk + 12));
      put4(bf1, 0,  *(const vf4*)(bp1 + kk));
      put4(bf1, 4,  *(const vf4*)(bp1 + kk + 4));
      put4(bf1, 8,  *(const vf4*)(bp1 + kk + 8));
      put4(bf1, 12, *(const vf4*)(bp1 + kk + 12));
    } else {
#pragma unroll
      for (int e = 0; e < 16; ++e) {
        int k = kk + half * 16 + e;
        bf0[e] = (__bf16)Bb[(long long)k * ldb + c0];
        bf1[e] = (__bf16)Bb[(long long)k * ldb + c1];
      }
    }
    acc0 = __builtin_amdgcn_wmma_f32_16x16x32_bf16(false, af, false, bf0, (short)0, acc0,
                                                   false, false);
    acc1 = __builtin_amdgcn_wmma_f32_16x16x32_bf16(false, af, false, bf1, (short)0, acc1,
                                                   false, false);
  }
  float bv0 = bias ? bias[c0] : 0.f;
  float bv1 = bias ? bias[c1] : 0.f;
#pragma unroll
  for (int r = 0; r < 8; ++r) {
    int orow = tm * 16 + r + half * 8;
    float v0 = acc0[r] + bv0;
    float v1 = acc1[r] + bv1;
    if (act == 1) { v0 = fmaxf(v0, 0.f); v1 = fmaxf(v1, 0.f); }
    Cb[(long long)orow * ldc + c0] = v0;
    Cb[(long long)orow * ldc + c1] = v1;
  }
}

// =====================================================================
// Two-pass WMMA MHA for dh=16, N=512: one workgroup per (batch, head).
// K/V are staged through LDS with GLOBAL_LOAD_ASYNC_TO_LDS_B128 (ASYNCcnt
// path), then converted to bf16: K as [key][d], V transposed [d][key], so
// every fragment gather is a contiguous ds_load_b128 (zero padding via
// address-select of a zeroed LDS row; no divergent loads).
// Pass 1: scores via WMMA, per-lane running max, ONE 16-lane butterfly
//         per q-tile to finalize row maxima.
// Pass 2: scores via WMMA again, P = exp(S - rowmax); O += P@V and
//         rowsum += P@ones, both via WMMA -- the ones-matmul makes every
//         lane's D element its own row sum, so no cross-lane reduction
//         and o[r]/l[r] line up exactly.
// q,k,v,ctx layout: [B, N, H] with head offset h*16.
// =====================================================================
__global__ __launch_bounds__(256) void attn_kernel(
    const float* __restrict__ Q, const float* __restrict__ Kt,
    const float* __restrict__ Vt, float* __restrict__ CTX)
{
  __shared__ alignas(16) __bf16 kl[NN * 16];        // [key][d]
  __shared__ alignas(16) __bf16 vlT[16 * NN];       // [d][key]
  __shared__ alignas(16) __bf16 pl[8][16 * 16];
  __shared__ alignas(16) __bf16 zpad[16];           // zero row for K-padding
  __shared__ alignas(16) float  stage[128 * 16];    // async-copy staging (8KB)

  int bh = blockIdx.x;
  int b = bh >> 2, h = bh & 3;
  const long long base = ((long long)b * NN) * HH + h * 16;

  if (threadIdx.x < 16) zpad[threadIdx.x] = (__bf16)0.f;

  // ---- stage K then V through LDS via async global->LDS copies ----
  for (int mat = 0; mat < 2; ++mat) {
    const float* src = mat ? Vt : Kt;
    for (int c = 0; c < 4; ++c) {                   // 128 keys / chunk
#pragma unroll
      for (int it = 0; it < 2; ++it) {              // 512 lanes-worth of b128
        int idx  = it * 256 + threadIdx.x;          // 0..511, 4 floats each
        int keyc = idx >> 2;                        // key within chunk
        int d4   = (idx & 3) * 4;
        unsigned lds = (unsigned)(unsigned long long)(void*)&stage[idx * 4];
        unsigned long long ga =
            (unsigned long long)(const void*)(src + base +
                (long long)(c * 128 + keyc) * HH + d4);
        asm volatile("global_load_async_to_lds_b128 %0, %1, off"
                     :: "v"(lds), "v"(ga) : "memory");
      }
      asm volatile("s_wait_asynccnt 0" ::: "memory");
      __syncthreads();
#pragma unroll
      for (int it = 0; it < 2; ++it) {
        int idx  = it * 256 + threadIdx.x;
        int key  = c * 128 + (idx >> 2);
        int d4   = (idx & 3) * 4;
        vf4 vv = *(const vf4*)&stage[idx * 4];
        if (mat == 0) {
#pragma unroll
          for (int j = 0; j < 4; ++j) kl[key * 16 + d4 + j] = (__bf16)vv[j];
        } else {
#pragma unroll
          for (int j = 0; j < 4; ++j) vlT[(d4 + j) * NN + key] = (__bf16)vv[j];
        }
      }
      __syncthreads();
    }
  }

  int wave = threadIdx.x >> 5, lane = threadIdx.x & 31;
  int half = lane >> 4, lr = lane & 15;
  const float scale = 0.25f;                  // 1/sqrt(16)

  v16bf onesb;
#pragma unroll
  for (int e = 0; e < 16; ++e) onesb[e] = (__bf16)1.0f;

  for (int qt = wave; qt < NN / 16; qt += 8) {
    v16bf qa;
    int qrow = qt * 16 + lr;
    {
      const float* qp = Q + base + (long long)qrow * HH;
      put4(qa, 0, *(const vf4*)(qp));
      put4(qa, 4, *(const vf4*)(qp + 4));
      put4(qa, 8, *(const vf4*)(qp + 8));
      put4(qa, 12, *(const vf4*)(qp + 12));
#pragma unroll
      for (int e = 0; e < 16; ++e) if (half) qa[e] = (__bf16)0.f;   // dh=16: upper lane half (K 16..31) zero-pad
    }

    // ---------------- pass 1: row maxima (no cross-lane work in loop) --------
    float m8[8];
#pragma unroll
    for (int r = 0; r < 8; ++r) m8[r] = -3.0e38f;
    for (int ct = 0; ct < NN / 16; ++ct) {
      const __bf16* ksrc = half ? zpad : &kl[(ct * 16 + lr) * 16];
      v8bf klo = *(const v8bf*)ksrc;
      v8bf khi = *(const v8bf*)(ksrc + 8);
      v16bf kb;
#pragma unroll
      for (int e = 0; e < 8; ++e) { kb[e] = klo[e]; kb[e + 8] = khi[e]; }
      v8f s = {};
      s = __builtin_amdgcn_wmma_f32_16x16x32_bf16(false, qa, false, kb, (short)0, s,
                                                  false, false);
#pragma unroll
      for (int r = 0; r < 8; ++r) m8[r] = fmaxf(m8[r], s[r]);
    }
    // one butterfly per q-tile finalizes row maxima across the 16 columns
#pragma unroll
    for (int r = 0; r < 8; ++r) {
      m8[r] *= scale;
#pragma unroll
      for (int off = 1; off < 16; off <<= 1) m8[r] = fmaxf(m8[r], __shfl_xor(m8[r], off));
    }

    // ---------------- pass 2: P = exp(S-m); O += P@V; l += P@ones ------------
    v8f o = {}, lacc = {};
    for (int ct = 0; ct < NN / 16; ++ct) {
      const __bf16* ksrc = half ? zpad : &kl[(ct * 16 + lr) * 16];
      v8bf klo = *(const v8bf*)ksrc;
      v8bf khi = *(const v8bf*)(ksrc + 8);
      v16bf kb;
#pragma unroll
      for (int e = 0; e < 8; ++e) { kb[e] = klo[e]; kb[e + 8] = khi[e]; }
      v8f s = {};
      s = __builtin_amdgcn_wmma_f32_16x16x32_bf16(false, qa, false, kb, (short)0, s,
                                                  false, false);
#pragma unroll
      for (int r = 0; r < 8; ++r) {
        float p = __expf(s[r] * scale - m8[r]);
        pl[wave][(r + half * 8) * 16 + lr] = (__bf16)p;   // C-layout -> LDS
      }
      // DS ops are in-order per wave; wait out the stores, fence the compiler.
      asm volatile("s_wait_dscnt 0" ::: "memory");
      // P A-frag: contiguous 16 bf16 per lane (2x b128) in lower half, upper lane half (K 16..31) zero.
      v16bf pa;
      {
        v8bf plo = *(const v8bf*)&pl[wave][lr * 16];
        v8bf phi = *(const v8bf*)&pl[wave][lr * 16 + 8];
#pragma unroll
        for (int e = 0; e < 8; ++e) { pa[e] = half ? (__bf16)0.f : plo[e]; pa[e + 8] = half ? (__bf16)0.f : phi[e]; }
      }
      // V B-frag: vlT[d][key] -> contiguous 16-key run; upper half -> zero row.
      const __bf16* vsrc = half ? zpad : &vlT[lr * NN + ct * 16];
      v8bf vlo = *(const v8bf*)vsrc;
      v8bf vhi = *(const v8bf*)(vsrc + 8);
      v16bf vb;
#pragma unroll
      for (int e = 0; e < 8; ++e) { vb[e] = vlo[e]; vb[e + 8] = vhi[e]; }

      o    = __builtin_amdgcn_wmma_f32_16x16x32_bf16(false, pa, false, vb, (short)0, o,
                                                     false, false);
      lacc = __builtin_amdgcn_wmma_f32_16x16x32_bf16(false, pa, false, onesb, (short)0,
                                                     lacc, false, false);
    }
#pragma unroll
    for (int r = 0; r < 8; ++r) {
      int orow = qt * 16 + r + half * 8;
      CTX[base + (long long)orow * HH + lr] = o[r] / lacc[r];
    }
  }
}

// ---------------- small elementwise / reduction kernels ----------------
__global__ void transpose_kernel(const float* __restrict__ in, float* __restrict__ out,
                                 int rows, int cols) {
  int i = blockIdx.x * blockDim.x + threadIdx.x;
  if (i >= rows * cols) return;
  int r = i / cols, c = i % cols;
  out[(long long)c * rows + r] = in[i];
}

__global__ void sigmoid_kernel(const float* in, float* out, int n) {
  int i = blockIdx.x * blockDim.x + threadIdx.x;
  if (i < n) out[i] = 1.f / (1.f + __expf(-in[i]));
}

__global__ void resmask_kernel(const float* __restrict__ xin, const float* __restrict__ att,
                               const float* __restrict__ mask, float* __restrict__ out,
                               long long n) {
  long long i = (long long)blockIdx.x * blockDim.x + threadIdx.x;
  if (i < n) out[i] = (xin[i] + 0.5f * att[i]) * mask[i & (NN * SS - 1)];
}

__global__ void mean_kernel(const float* __restrict__ x, float* __restrict__ out, int n) {
  int i = blockIdx.x * blockDim.x + threadIdx.x;
  if (i >= n) return;
  float s = 0.f;
  for (int b = 0; b < BB; ++b) s += x[(long long)b * NN * SS + i];
  out[i] = s * (1.f / BB);
}

__global__ void stack_kernel(const float* b0, const float* b1, float* out, int L) {
  int i = blockIdx.x * blockDim.x + threadIdx.x;
  if (i >= NN * L * SS) return;
  int n = i / (L * SS), rem = i % (L * SS), l = rem / SS, s = rem % SS;
  const float* src = (l == 0) ? b0 : b1;
  out[i] = src[n * SS + s];
}

__global__ void tattn_kernel(const float* __restrict__ q, const float* __restrict__ k,
                             const float* __restrict__ v, float* __restrict__ ctx, int L) {
  int idx = blockIdx.x * blockDim.x + threadIdx.x;
  if (idx >= NN * 2) return;
  int n = idx >> 1, h = idx & 1;
  const float scale = 0.17677669529663687f;   // 1/sqrt(32)
  for (int tq = 0; tq < L; ++tq) {
    float sc[5]; float mx = -3e38f;
    for (int tk = 0; tk < L; ++tk) {
      float s = 0.f;
      for (int d = 0; d < 32; ++d)
        s += q[(long long)(n * L + tq) * HH + h * 32 + d] *
             k[(long long)(n * L + tk) * HH + h * 32 + d];
      sc[tk] = s * scale; mx = fmaxf(mx, sc[tk]);
    }
    float se = 0.f;
    for (int tk = 0; tk < L; ++tk) { sc[tk] = __expf(sc[tk] - mx); se += sc[tk]; }
    for (int d = 0; d < 32; ++d) {
      float a = 0.f;
      for (int tk = 0; tk < L; ++tk)
        a += sc[tk] * v[(long long)(n * L + tk) * HH + h * 32 + d];
      ctx[(long long)(n * L + tq) * HH + h * 32 + d] = a / se;
    }
  }
}

__global__ void wsum_kernel(const float* __restrict__ src, float* __restrict__ out, int L) {
  int i = blockIdx.x * blockDim.x + threadIdx.x;
  if (i >= NN * SS) return;
  int n = i / SS, s = i % SS;
  float acc = 0.f, den = 0.f;
  for (int l = 0; l < L; ++l) {
    float w = powf(0.9f, (float)(L - 1 - l));
    acc += w * src[(long long)(n * L + l) * SS + s];
    den += w;
  }
  out[i] = acc / den;
}

__global__ void combine_kernel(float* st, const float* recon) {
  __shared__ float red[256];
  int row = blockIdx.x, t = threadIdx.x;
  float x = st[row * SS + t];
  red[t] = x * x;
  __syncthreads();
  for (int s = 128; s > 0; s >>= 1) { if (t < s) red[t] += red[t + s]; __syncthreads(); }
  float imp = sqrtf(red[0]);
  st[row * SS + t] = (imp > 0.f) ? recon[row * SS + t] : x;
}

__global__ void topk_kernel(const float* st, float* out) {
  __shared__ float a[256];
  __shared__ float kred[256];
  int row = blockIdx.x, t = threadIdx.x;
  float val = st[row * SS + t];
  float av = fabsf(val);
  a[t] = av;
  __syncthreads();
  int gt = 0;
  for (int j = 0; j < 256; ++j) gt += (a[j] > av) ? 1 : 0;
  kred[t] = (gt < 32) ? av : 3e38f;           // kth-largest = min{v : count(>v) < K}
  __syncthreads();
  for (int s = 128; s > 0; s >>= 1) { if (t < s) kred[t] = fminf(kred[t], kred[t + s]); __syncthreads(); }
  float kth = kred[0];
  out[row * SS + t] = (av >= kth) ? val : 0.f;
}

// =====================================================================
extern "C" void kernel_launch(void* const* d_in, const int* in_sizes, int n_in,
                              void* d_out, int out_size, void* d_ws, size_t ws_size,
                              hipStream_t stream) {
  (void)in_sizes; (void)n_in; (void)out_size; (void)ws_size;
  const float* x       = (const float*)d_in[0];
  const float* ent     = (const float*)d_in[1];
  const float* mixing  = (const float*)d_in[2];
  const float* readout = (const float*)d_in[3];
  const float* n_qw = (const float*)d_in[4];  const float* n_qb = (const float*)d_in[5];
  const float* n_kw = (const float*)d_in[6];  const float* n_kb = (const float*)d_in[7];
  const float* n_vw = (const float*)d_in[8];  const float* n_vb = (const float*)d_in[9];
  const float* n_ow = (const float*)d_in[10]; const float* n_ob = (const float*)d_in[11];
  const float* t_qw = (const float*)d_in[12]; const float* t_qb = (const float*)d_in[13];
  const float* t_kw = (const float*)d_in[14]; const float* t_kb = (const float*)d_in[15];
  const float* t_vw = (const float*)d_in[16]; const float* t_vb = (const float*)d_in[17];
  const float* t_ow = (const float*)d_in[18]; const float* t_ob = (const float*)d_in[19];
  const float* enc_w = (const float*)d_in[20]; const float* enc_b = (const float*)d_in[21];
  const float* dec_w = (const float*)d_in[22]; const float* dec_b = (const float*)d_in[23];
  float* outp = (float*)d_out;

  const long long BN  = (long long)BB * NN;         // 32768
  const long long BNS = BN * SS;                    // 8388608
  const long long BNH = BN * HH;                    // 2097152
  const int NS = NN * SS;                           // 131072

  float* W = (float*)d_ws;
  size_t off = 0;
  auto alloc = [&](size_t n) { float* p = W + off; off += n; return p; };
  float* XA    = alloc(BNS);
  float* XB    = alloc(BNS);
  float* TMP   = alloc(BNS);
  float* Qb    = alloc(BNH);
  float* Kb    = alloc(BNH);
  float* Vb    = alloc(BNH);
  float* CTX   = alloc(BNH);
  float* MASK  = alloc(NS);
  float* BUF0  = alloc(NS);
  float* BUF1  = alloc(NS);
  float* BSTK  = alloc((size_t)NN * 2 * SS);
  float* TQ    = alloc((size_t)NN * 2 * HH);
  float* TK    = alloc((size_t)NN * 2 * HH);
  float* TV    = alloc((size_t)NN * 2 * HH);
  float* TCTX  = alloc((size_t)NN * 2 * HH);
  float* TATT  = alloc((size_t)NN * 2 * SS);
  float* SW    = alloc(NS);
  float* HID   = alloc((size_t)NN * CC_);
  float* RECON = alloc(NS);
  // pre-transposed operands (BT layout [Ncol, K]; MIXT = mixing^T for A side)
  float* MIXT  = alloc((size_t)NN * NN);
  float* QWT   = alloc((size_t)HH * SS);
  float* KWT   = alloc((size_t)HH * SS);
  float* VWT   = alloc((size_t)HH * SS);
  float* OWT   = alloc((size_t)SS * HH);
  float* TQWT  = alloc((size_t)HH * SS);
  float* TKWT  = alloc((size_t)HH * SS);
  float* TVWT  = alloc((size_t)HH * SS);
  float* TOWT  = alloc((size_t)SS * HH);
  float* ENCT  = alloc((size_t)CC_ * SS);
  float* DECT  = alloc((size_t)SS * CC_);

  auto tr = [&](const float* in, float* out, int rows, int cols) {
    transpose_kernel<<<(rows * cols + 255) / 256, 256, 0, stream>>>(in, out, rows, cols);
  };
  auto gemm = [&](const float* A, int lda, const float* Bm, int ldb, int tBT,
                  const float* bias, float* Cm, int ldc, int M, int Ncol, int K,
                  int batch, long long sA, long long sB, long long sC, int act) {
    int tiles = (M >> 4) * (Ncol >> 5);
    dim3 g((tiles + 7) / 8, batch);
    if (tBT)
      gemm_wmma<1><<<g, 256, 0, stream>>>(A, lda, Bm, ldb, bias, Cm, ldc, M, Ncol, K,
                                          sA, sB, sC, act);
    else
      gemm_wmma<0><<<g, 256, 0, stream>>>(A, lda, Bm, ldb, bias, Cm, ldc, M, Ncol, K,
                                          sA, sB, sC, act);
  };

  // one-time operand prep
  sigmoid_kernel<<<(NS + 255) / 256, 256, 0, stream>>>(ent, MASK, NS);
  tr(mixing, MIXT, NN, NN);
  tr(n_qw, QWT, SS, HH);  tr(n_kw, KWT, SS, HH);  tr(n_vw, VWT, SS, HH);
  tr(n_ow, OWT, HH, SS);
  tr(t_qw, TQWT, SS, HH); tr(t_kw, TKWT, SS, HH); tr(t_vw, TVWT, SS, HH);
  tr(t_ow, TOWT, HH, SS);
  tr(enc_w, ENCT, SS, CC_); tr(dec_w, DECT, CC_, SS);

  for (int layer = 0; layer < 2; ++layer) {
    const float* xin = (layer == 0) ? x : XB;
    // ---- spatial MHA: QKV projections, attention, out-proj ----
    gemm(xin, SS, QWT, SS, 1, n_qb, Qb, HH, (int)BN, HH, SS, 1, 0, 0, 0, 0);
    gemm(xin, SS, KWT, SS, 1, n_kb, Kb, HH, (int)BN, HH, SS, 1, 0, 0, 0, 0);
    gemm(xin, SS, VWT, SS, 1, n_vb, Vb, HH, (int)BN, HH, SS, 1, 0, 0, 0, 0);
    attn_kernel<<<BB * 4, 256, 0, stream>>>(Qb, Kb, Vb, CTX);
    gemm(CTX, HH, OWT, HH, 1, n_ob, TMP, SS, (int)BN, SS, HH, 1, 0, 0, 0, 0);
    // ---- residual + entanglement mask ----
    resmask_kernel<<<(unsigned)(BNS / 256), 256, 0, stream>>>(xin, TMP, MASK, XA, BNS);
    // ---- mixing: x[b,m,s] = sum_n mixing[n,m]*x[b,n,s]; A=mixing^T, B normal-layout ----
    gemm(MIXT, NN, XA, SS, 0, nullptr, XB, SS, NN, SS, NN, BB,
         0, (long long)NN * SS, (long long)NN * SS, 0);
    // ---- states = mean over batch; push into buffer ----
    mean_kernel<<<(NS + 255) / 256, 256, 0, stream>>>(XB, (layer == 0) ? BUF0 : BUF1, NS);
    int L = layer + 1;
    stack_kernel<<<(NN * L * SS + 255) / 256, 256, 0, stream>>>(BUF0, BUF1, BSTK, L);
    const float* src = BSTK;
    if (L > 1) {
      int Mrows = NN * L;
      gemm(BSTK, SS, TQWT, SS, 1, t_qb, TQ, HH, Mrows, HH, SS, 1, 0, 0, 0, 0);
      gemm(BSTK, SS, TKWT, SS, 1, t_kb, TK, HH, Mrows, HH, SS, 1, 0, 0, 0, 0);
      gemm(BSTK, SS, TVWT, SS, 1, t_vb, TV, HH, Mrows, HH, SS, 1, 0, 0, 0, 0);
      tattn_kernel<<<(NN * 2 + 255) / 256, 256, 0, stream>>>(TQ, TK, TV, TCTX, L);
      gemm(TCTX, HH, TOWT, HH, 1, t_ob, TATT, SS, Mrows, SS, HH, 1, 0, 0, 0, 0);
      src = TATT;
    }
    // ---- recency-weighted buffer collapse ----
    wsum_kernel<<<(NS + 255) / 256, 256, 0, stream>>>(src, SW, L);
    // ---- encoder(relu)/decoder reconstruction + importance gate ----
    gemm(SW, SS, ENCT, SS, 1, enc_b, HID, CC_, NN, CC_, SS, 1, 0, 0, 0, 1);
    gemm(HID, CC_, DECT, CC_, 1, dec_b, RECON, SS, NN, SS, CC_, 1, 0, 0, 0, 0);
    combine_kernel<<<NN, 256, 0, stream>>>(SW, RECON);
    // ---- top-K sparsification (last layer writes straight to d_out) ----
    float* st_out = (layer == 1) ? (outp + BNS) : SW;
    topk_kernel<<<NN, 256, 0, stream>>>(SW, st_out);
  }
  // ---- readout: out = x @ readout_w^T (readout_w is already BT layout) ----
  gemm(XB, SS, readout, SS, 1, nullptr, outp, SS, (int)BN, SS, SS, 1, 0, 0, 0, 0);
}
